// RNNEncoder_75917841924237
// MI455X (gfx1250) — matrix operand order repaired and checked by reading, not compiled, not run on hardware
//
#include <hip/hip_runtime.h>
#include <hip/hip_bf16.h>

// =====================================================================
// LayerNorm-LSTM bidirectional encoder for MI455X (gfx1250, wave32).
//
// Strategy (latency-dominated recurrence on a tensor-op chip):
//   1) Precompute ig = LN(x @ W_ih^T + b_ih)*g + be for ALL timesteps as a
//      big parallel f16 WMMA GEMM (grid-wide, 1024 blocks/dir).
//   2) Persistent recurrent kernel: 1 workgroup (32 waves) per direction.
//      Per step: hg = LN(h @ W_hh^T + b_hh) via v_wmma_f32_16x16x32_f16
//      (h lives in LDS as f16), LN row-stats via shuffle + ds_add_f32,
//      gates staged in LDS f16, c kept in registers, LN over c via
//      intra-16-lane shuffles only. 4 barriers/step.
//      NEW: ig[t] tile (128 KB) is DMA'd global->LDS each step with
//      global_load_async_to_lds_b128 (ASYNCcnt), fully overlapped with
//      the WMMA GEMM, drained with s_wait_asynccnt before use.
//
// d_in flattening assumption (jax pytree, dict keys sorted):
//   d_in[0]=x [B,T,D] f32, d_in[1]=mask (unused), then per (layer,dir):
//   [W_hh, W_ih, b_hh, b_ih, be_hh, be_ho, be_ih, g_hh, g_ho, g_ih]
// Workspace requirement: ~98 MB.
// =====================================================================

typedef __attribute__((ext_vector_type(16))) _Float16 v16h;
typedef __attribute__((ext_vector_type(8)))  _Float16 v8h;
typedef __attribute__((ext_vector_type(8)))  float    v8f;

#define TSEQ 256
#define NB   64
#define HDIM 256
#define G4   1024
#define MROWS (TSEQ * NB)          // 16384
#define LN_EPS 1e-5f

__device__ __forceinline__ float sigf(float x) {
  return 1.0f / (1.0f + __expf(-x));
}
__device__ __forceinline__ float tanh_fast(float x) {
  float e = __expf(-2.0f * fabsf(x));
  float r = (1.0f - e) / (1.0f + e);
  return copysignf(r, x);
}

// ---------------- x: [B,T,D] f32 -> [T,B,D] f16 -------------------------
__global__ void k_transpose_cast(const float* __restrict__ x,
                                 _Float16* __restrict__ y,
                                 int B, int T, int D) {
  size_t i = (size_t)blockIdx.x * blockDim.x + threadIdx.x;
  size_t total = (size_t)B * T * D;
  if (i >= total) return;
  int d = (int)(i % D);
  size_t bt = i / D;
  int t = (int)(bt % T);
  int b = (int)(bt / T);
  y[((size_t)t * B + b) * D + d] = (_Float16)x[i];
}

// ---------------- generic f32 -> f16 cast -------------------------------
__global__ void k_cast_f16(const float* __restrict__ x,
                           _Float16* __restrict__ y, int n) {
  int i = blockIdx.x * blockDim.x + threadIdx.x;
  if (i < n) y[i] = (_Float16)x[i];
}

// ---------------- ig = f16( LN(X @ W^T + b) * g + be ) ------------------
// X: [MROWS x K] f16 row-major. W: [1024 x K] f16 row-major.
// One block = 16 output rows x 1024 cols. 32 waves, 2 N-tiles each.
__global__ __launch_bounds__(1024) void k_ih_ln(
    const _Float16* __restrict__ X, const _Float16* __restrict__ W,
    const float* __restrict__ bias, const float* __restrict__ gain,
    const float* __restrict__ beta, _Float16* __restrict__ IG, int K) {
  __shared__ float rsum[16], rsumsq[16], rmean[16], rrstd[16];
  int tid = threadIdx.x;
  int w = tid >> 5, lane = tid & 31;
  int hi = lane >> 4, l15 = lane & 15;
  size_t row0 = (size_t)blockIdx.x * 16;
  if (tid < 16) { rsum[tid] = 0.f; rsumsq[tid] = 0.f; }
  __syncthreads();

  v8f acc0 = {}, acc1 = {};
  const _Float16* ap = X + (row0 + l15) * K + (hi ? 16 : 0);
  int n0 = (2 * w + 0) * 16 + l15, n1 = n0 + 16;
  const _Float16* bp0 = W + (size_t)n0 * K + (hi ? 16 : 0);
  const _Float16* bp1 = W + (size_t)n1 * K + (hi ? 16 : 0);
  for (int kb = 0; kb < K; kb += 32) {
    v8h alo = *(const v8h*)(ap + kb);
    v8h ahi = *(const v8h*)(ap + kb + 8);
    v16h a = __builtin_shufflevector(alo, ahi, 0, 1, 2, 3, 4, 5, 6, 7, 8, 9,
                                     10, 11, 12, 13, 14, 15);
    v16h b0 = *(const v16h*)(bp0 + kb);
    v16h b1 = *(const v16h*)(bp1 + kb);
    acc0 = __builtin_amdgcn_wmma_f32_16x16x32_f16(false, a, false, b0,
                                                  (short)0, acc0, false, false);
    acc1 = __builtin_amdgcn_wmma_f32_16x16x32_f16(false, a, false, b1,
                                                  (short)0, acc1, false, false);
  }
  float bn0 = bias[n0], bn1 = bias[n1];
#pragma unroll
  for (int r = 0; r < 8; ++r) {
    float v0 = acc0[r] + bn0, v1 = acc1[r] + bn1;
    float s1 = v0 + v1, s2 = v0 * v0 + v1 * v1;
#pragma unroll
    for (int m = 1; m < 16; m <<= 1) {
      s1 += __shfl_xor(s1, m, 32);
      s2 += __shfl_xor(s2, m, 32);
    }
    if (l15 == 0) {
      int ml = r + hi * 8;
      atomicAdd(&rsum[ml], s1);
      atomicAdd(&rsumsq[ml], s2);
    }
    acc0[r] = v0;
    acc1[r] = v1;
  }
  __syncthreads();
  if (tid < 16) {
    float mu = rsum[tid] * (1.f / 1024.f);
    float var = rsumsq[tid] * (1.f / 1024.f) - mu * mu;
    rmean[tid] = mu;
    rrstd[tid] = rsqrtf(var + LN_EPS);
  }
  __syncthreads();
  float g0 = gain[n0], g1 = gain[n1];
  float e0 = beta[n0], e1 = beta[n1];
#pragma unroll
  for (int r = 0; r < 8; ++r) {
    int ml = r + hi * 8;
    float mu = rmean[ml], rs = rrstd[ml];
    size_t orow = (row0 + ml) * (size_t)G4;
    IG[orow + n0] = (_Float16)((acc0[r] - mu) * rs * g0 + e0);
    IG[orow + n1] = (_Float16)((acc1[r] - mu) * rs * g1 + e1);
  }
}

// ---------------- persistent recurrent kernel ---------------------------
struct RecArgs {
  const _Float16* ig[2];
  const _Float16* whh[2];
  const float* b_hh[2];
  const float* g_hh[2];
  const float* be_hh[2];
  const float* g_ho[2];
  const float* be_ho[2];
};

// Dynamic LDS: hbuf 64x256 f16 | gbuf 64x1024 f16 | igbuf 64x1024 f16 |
//              4x64 f32 stats  = 296 KB (<= 320 KB WGP LDS, 1 WG/WGP)
#define REC_LDS_BYTES (NB * HDIM * 2 + 2 * NB * G4 * 2 + 4 * NB * 4)

extern __shared__ char smem_dyn[];

__global__ __launch_bounds__(1024) void k_rec(RecArgs P,
                                              _Float16* __restrict__ outF16,
                                              float* __restrict__ outF32,
                                              int T, int final_layer) {
  int dir = blockIdx.x;  // 0 = forward, 1 = backward
  const _Float16* IG = P.ig[dir];
  const _Float16* WHH = P.whh[dir];
  const float* bhhp = P.b_hh[dir];
  const float* ghhp = P.g_hh[dir];
  const float* ehhp = P.be_hh[dir];
  const float* ghop = P.g_ho[dir];
  const float* ehop = P.be_ho[dir];

  _Float16* hbuf = (_Float16*)smem_dyn;                          // 64*256
  _Float16* gbuf = (_Float16*)(smem_dyn + NB * HDIM * 2);        // 64*1024
  _Float16* igbuf = (_Float16*)(smem_dyn + NB * HDIM * 2 + NB * G4 * 2);
  float* rsum = (float*)(smem_dyn + NB * HDIM * 2 + 2 * NB * G4 * 2);
  float* rsumsq = rsum + NB;
  float* rmean = rsum + 2 * NB;
  float* rrstd = rsum + 3 * NB;

  int tid = threadIdx.x;
  int w = tid >> 5, lane = tid & 31;
  int hi = lane >> 4, l15 = lane & 15;
  int Mb = w >> 3;  // 0..3  (rows Mb*16 .. +15)
  int Ng = w & 7;   // 0..7  (N-tiles Ng*8 .. +7)

  // zero h state
  for (int i = tid; i < NB * HDIM; i += 1024) hbuf[i] = (_Float16)0.f;

  // per-thread ownership for elementwise phase: batch row + 16 j values
  int b_own = tid >> 4;
  int jbase = (tid & 15) * 16;
  float c[16], ogate[16];
#pragma unroll
  for (int i = 0; i < 16; ++i) c[i] = 0.f;

  // step-invariant preloads
  int nvals[8];
  float bh[8], gh[8], eh[8];
  const _Float16* wp[8];
#pragma unroll
  for (int nt = 0; nt < 8; ++nt) {
    int n = (Ng * 8 + nt) * 16 + l15;
    nvals[nt] = n;
    bh[nt] = bhhp[n];
    gh[nt] = ghhp[n];
    eh[nt] = ehhp[n];
    wp[nt] = WHH + (size_t)n * HDIM + (hi ? 16 : 0);
  }
  float gho[16], eho[16];
#pragma unroll
  for (int i = 0; i < 16; ++i) {
    gho[i] = ghop[jbase + i];
    eho[i] = ehop[jbase + i];
  }
  const _Float16* hp = hbuf + (Mb * 16 + l15) * HDIM + (hi ? 16 : 0);
  // async-DMA base addresses for ig tile (thread -> 8 x b128 chunks)
  unsigned ig_lds_base = (unsigned)(uintptr_t)(igbuf + tid * 8);
  __syncthreads();

  for (int st = 0; st < T; ++st) {
    int t = dir ? (T - 1 - st) : st;
    if (tid < NB) { rsum[tid] = 0.f; rsumsq[tid] = 0.f; }
    __syncthreads();  // B1: h ready, stats zeroed, gbuf/igbuf free

    // ---- async DMA: ig[t] (64x1024 f16 = 128 KB) global -> LDS ----
    // Overlaps with the WMMA GEMM below; tracked by ASYNCcnt.
    {
      const _Float16* gsrc = IG + (size_t)t * NB * G4 + (size_t)tid * 8;
#pragma unroll
      for (int i = 0; i < 8; ++i) {
        unsigned long long ga =
            (unsigned long long)(uintptr_t)(gsrc + (size_t)i * 8192);
        unsigned la = ig_lds_base + (unsigned)i * 16384u;
        asm volatile("global_load_async_to_lds_b128 %0, %1, off"
                     :
                     : "v"(la), "v"(ga)
                     : "memory");
      }
    }

    // ---- hg_pre = h @ W_hh^T  (f16 WMMA, f32 acc) ----
    v8f acc[8] = {};
    for (int kb = 0; kb < HDIM; kb += 32) {
      v8h alo = *(const v8h*)(hp + kb);
      v8h ahi = *(const v8h*)(hp + kb + 8);
      v16h a = __builtin_shufflevector(alo, ahi, 0, 1, 2, 3, 4, 5, 6, 7, 8, 9,
                                       10, 11, 12, 13, 14, 15);
#pragma unroll
      for (int nt = 0; nt < 8; ++nt) {
        v16h bfr = *(const v16h*)(wp[nt] + kb);
        acc[nt] = __builtin_amdgcn_wmma_f32_16x16x32_f16(
            false, a, false, bfr, (short)0, acc[nt], false, false);
      }
    }

    // ---- LN row stats over 1024 gate cols (shuffle + ds_add_f32) ----
#pragma unroll
    for (int r = 0; r < 8; ++r) {
      float s1 = 0.f, s2 = 0.f;
#pragma unroll
      for (int nt = 0; nt < 8; ++nt) {
        float v = acc[nt][r] + bh[nt];
        s1 += v;
        s2 += v * v;
        acc[nt][r] = v;
      }
#pragma unroll
      for (int m = 1; m < 16; m <<= 1) {
        s1 += __shfl_xor(s1, m, 32);
        s2 += __shfl_xor(s2, m, 32);
      }
      if (l15 == 0) {
        int row = Mb * 16 + r + hi * 8;
        atomicAdd(&rsum[row], s1);
        atomicAdd(&rsumsq[row], s2);
      }
    }
    // drain this thread's async ig transfers before the barrier so the
    // whole tile is LDS-visible to everyone after B2
    asm volatile("s_wait_asynccnt 0" ::: "memory");
    __syncthreads();  // B2: stats complete, ig tile resident in LDS
    if (tid < NB) {
      float mu = rsum[tid] * (1.f / 1024.f);
      float var = rsumsq[tid] * (1.f / 1024.f) - mu * mu;
      rmean[tid] = mu;
      rrstd[tid] = rsqrtf(var + LN_EPS);
    }
    __syncthreads();  // B3: mean/rstd ready

    // ---- gates = LN(hg)*g+be + ig[t]  -> gbuf (f16) ----
#pragma unroll
    for (int r = 0; r < 8; ++r) {
      int row = Mb * 16 + r + hi * 8;
      float mu = rmean[row], rs = rrstd[row];
#pragma unroll
      for (int nt = 0; nt < 8; ++nt) {
        int n = nvals[nt];
        float hg = (acc[nt][r] - mu) * rs * gh[nt] + eh[nt];
        float igv = (float)igbuf[(size_t)row * G4 + n];
        gbuf[row * G4 + n] = (_Float16)(hg + igv);
      }
    }
    __syncthreads();  // B4: gbuf ready

    // ---- cell / hidden update + LN over c (256) ----
    _Float16* grow = gbuf + b_own * G4 + jbase;
    float cs = 0.f, cs2 = 0.f;
#pragma unroll
    for (int i = 0; i < 16; ++i) {
      float gi = (float)grow[i];
      float gf = (float)grow[HDIM + i];
      float gg = (float)grow[2 * HDIM + i];
      float go = (float)grow[3 * HDIM + i];
      c[i] = sigf(gf) * c[i] + sigf(gi) * tanh_fast(gg);
      ogate[i] = go;
      cs += c[i];
      cs2 += c[i] * c[i];
    }
#pragma unroll
    for (int m = 1; m < 16; m <<= 1) {
      cs += __shfl_xor(cs, m, 32);
      cs2 += __shfl_xor(cs2, m, 32);
    }
    float cmu = cs * (1.f / 256.f);
    float cvar = cs2 * (1.f / 256.f) - cmu * cmu;
    float crs = rsqrtf(cvar + LN_EPS);
#pragma unroll
    for (int i = 0; i < 16; ++i) {
      float cn = (c[i] - cmu) * crs * gho[i] + eho[i];
      float hv = sigf(ogate[i]) * tanh_fast(cn);
      hbuf[b_own * HDIM + jbase + i] = (_Float16)hv;
      if (!final_layer) {
        outF16[((size_t)t * NB + b_own) * (2 * HDIM) + dir * HDIM + jbase + i] =
            (_Float16)hv;
      } else if (t == T - 1) {
        outF32[b_own * (2 * HDIM) + dir * HDIM + jbase + i] = hv;
      }
    }
    // next iteration's B1 orders h/gbuf/igbuf reuse
  }
}

// =====================================================================
extern "C" void kernel_launch(void* const* d_in, const int* in_sizes, int n_in,
                              void* d_out, int out_size, void* d_ws,
                              size_t ws_size, hipStream_t stream) {
  (void)in_sizes; (void)n_in; (void)out_size; (void)ws_size;
  const int B = NB, T = TSEQ, D = 256;
  const size_t M = (size_t)MROWS;

  const float* X = (const float*)d_in[0];
  // params: per (layer,dir) sorted-key order
  const float *W_hh[2][2], *W_ih[2][2], *b_hh[2][2], *b_ih[2][2];
  const float *be_hh[2][2], *be_ho[2][2], *be_ih[2][2];
  const float *g_hh[2][2], *g_ho[2][2], *g_ih[2][2];
  for (int l = 0; l < 2; ++l)
    for (int d = 0; d < 2; ++d) {
      int base = 2 + (l * 2 + d) * 10;
      W_hh[l][d] = (const float*)d_in[base + 0];
      W_ih[l][d] = (const float*)d_in[base + 1];
      b_hh[l][d] = (const float*)d_in[base + 2];
      b_ih[l][d] = (const float*)d_in[base + 3];
      be_hh[l][d] = (const float*)d_in[base + 4];
      be_ho[l][d] = (const float*)d_in[base + 5];
      be_ih[l][d] = (const float*)d_in[base + 6];
      g_hh[l][d] = (const float*)d_in[base + 7];
      g_ho[l][d] = (const float*)d_in[base + 8];
      g_ih[l][d] = (const float*)d_in[base + 9];
    }

  // workspace carve (~98 MB)
  char* wsp = (char*)d_ws;
  auto take = [&](size_t bytes) -> char* {
    char* p = wsp;
    wsp += (bytes + 255) & ~(size_t)255;
    return p;
  };
  _Float16* x0 = (_Float16*)take(M * 256 * 2);   // layer0 input [T,B,256] f16
  _Float16* x1 = (_Float16*)take(M * 512 * 2);   // layer1 input [T,B,512] f16
  _Float16 *wihf[2][2], *whhf[2][2];
  for (int l = 0; l < 2; ++l)
    for (int d = 0; d < 2; ++d) {
      wihf[l][d] = (_Float16*)take((size_t)G4 * (l ? 512 : 256) * 2);
      whhf[l][d] = (_Float16*)take((size_t)G4 * HDIM * 2);
    }
  _Float16* igb[2];
  igb[0] = (_Float16*)take(M * G4 * 2);
  igb[1] = (_Float16*)take(M * G4 * 2);

  // 1) transpose+cast x
  {
    size_t n = (size_t)B * T * D;
    k_transpose_cast<<<(unsigned)((n + 255) / 256), 256, 0, stream>>>(X, x0, B,
                                                                      T, D);
  }
  // 2) cast weights to f16
  for (int l = 0; l < 2; ++l)
    for (int d = 0; d < 2; ++d) {
      int nih = G4 * (l ? 512 : 256);
      int nhh = G4 * HDIM;
      k_cast_f16<<<(nih + 255) / 256, 256, 0, stream>>>(W_ih[l][d], wihf[l][d],
                                                        nih);
      k_cast_f16<<<(nhh + 255) / 256, 256, 0, stream>>>(W_hh[l][d], whhf[l][d],
                                                        nhh);
    }

  for (int l = 0; l < 2; ++l) {
    const _Float16* xin = l ? x1 : x0;
    int K = l ? 512 : 256;
    // 3) parallel ih GEMM + LN per direction
    for (int d = 0; d < 2; ++d) {
      k_ih_ln<<<MROWS / 16, 1024, 0, stream>>>(xin, wihf[l][d], b_ih[l][d],
                                               g_ih[l][d], be_ih[l][d], igb[d],
                                               K);
    }
    // 4) persistent recurrence, both directions concurrently
    RecArgs ra;
    for (int d = 0; d < 2; ++d) {
      ra.ig[d] = igb[d];
      ra.whh[d] = whhf[l][d];
      ra.b_hh[d] = b_hh[l][d];
      ra.g_hh[d] = g_hh[l][d];
      ra.be_hh[d] = be_hh[l][d];
      ra.g_ho[d] = g_ho[l][d];
      ra.be_ho[d] = be_ho[l][d];
    }
    k_rec<<<2, 1024, REC_LDS_BYTES, stream>>>(ra, x1, (float*)d_out, T,
                                              /*final_layer=*/l == 1);
  }
}
